// hashNerf_35330400977258
// MI455X (gfx1250) — matrix operand order repaired and checked by reading, not compiled, not run on hardware
//
#include <hip/hip_runtime.h>

// CDNA5 / gfx1250 fused HashNeRF forward:
//   hash-grid bilinear encoding (degenerate %2 hash -> 16-entry table) +
//   32->64->64->64->3 MLP via v_wmma_f32_16x16x32_f16.
// Each wave owns TWO 16-point M-tiles (32 points): B fragments are loaded
// once and reused across both M-tiles, and the two WMMAs per B-fragment are
// independent (better XDL pipelining). Bias is folded into the WMMA C
// operand; leaky-ReLU is fmax(z, 0.01z).

typedef __attribute__((ext_vector_type(16))) _Float16 v16h;
typedef __attribute__((ext_vector_type(8)))  _Float16 v8h;
typedef __attribute__((ext_vector_type(8)))  float    v8f;

#define T_SIZE 262144
#define PTS_PER_WAVE 32
#define PTS_PER_BLOCK 256   // 8 waves * 32 points

// Build a 16-half WMMA fragment from two contiguous 16-byte LDS chunks.
static __device__ __forceinline__ v16h frag_from_lds(const _Float16* p0,
                                                     const _Float16* p1) {
  v8h lo = *(const v8h*)p0;
  v8h hi = *(const v8h*)p1;
  return __builtin_shufflevector(lo, hi, 0,1,2,3,4,5,6,7,8,9,10,11,12,13,14,15);
}

static __device__ __forceinline__ v8f wmma_f16(v16h a, v16h b, v8f c) {
  // (neg_a, A, neg_b, B, c_mod, C, reuse_a, reuse_b)
  return __builtin_amdgcn_wmma_f32_16x16x32_f16(false, a, false, b,
                                                (short)0, c, false, false);
}

static __device__ __forceinline__ v8f splat8(float x) {
  v8f v = {x, x, x, x, x, x, x, x};
  return v;
}

static __device__ __forceinline__ float lrelu(float z) {
  return fmaxf(z, 0.01f * z);   // == leaky relu for all z
}

static __device__ __forceinline__ void lds_fence() {
  // Same-wave LDS ops are processed in order; this is a compiler reorder
  // fence plus an explicit CDNA5 dscnt drain before cross-lane RAW reads.
  asm volatile("s_wait_dscnt 0" ::: "memory");
}

// One hidden layer for a 32x64 activation panel: act @ W(64x64) + b, leaky
// relu, in-place in sA (rows 0..31, row stride 64 halves).
// sWt is the transposed [n][k] f16 weight tile (row stride 64).
static __device__ __forceinline__ void mlp_layer64(_Float16* sA,
                                                   const _Float16* sWt,
                                                   const float* sB,
                                                   int mrow, int seg) {
  const _Float16* aRow0 = sA + mrow * 64;          // M-tile 0: rows 0..15
  const _Float16* aRow1 = sA + (mrow + 16) * 64;   // M-tile 1: rows 16..31
  const int kb = seg * 8;
  // A fragments for K=0..31 and K=32..63 (ISA 16-bit A 16x32 layout).
  v16h a00 = frag_from_lds(aRow0 + kb,      aRow0 + kb + 16);
  v16h a01 = frag_from_lds(aRow0 + 32 + kb, aRow0 + 32 + kb + 16);
  v16h a10 = frag_from_lds(aRow1 + kb,      aRow1 + kb + 16);
  v16h a11 = frag_from_lds(aRow1 + 32 + kb, aRow1 + 32 + kb + 16);
#pragma unroll
  for (int t = 0; t < 4; ++t) {
    const _Float16* bp = sWt + (t * 16 + mrow) * 64 + seg * 8;
    v16h b0 = frag_from_lds(bp,      bp + 16);  // K = seg*8 + {0..7, 16..23}
    v16h b1 = frag_from_lds(bp + 32, bp + 48);  // K = 32 + seg*8 + {0..7, 16..23}
    const float bias = sB[t * 16 + mrow];
    v8f c0 = splat8(bias);                      // fold bias into C operand
    v8f c1 = splat8(bias);
    c0 = wmma_f16(a00, b0, c0);
    c1 = wmma_f16(a10, b0, c1);                 // independent of c0 chain
    c0 = wmma_f16(a01, b1, c0);
    c1 = wmma_f16(a11, b1, c1);
#pragma unroll
    for (int r = 0; r < 8; ++r) {               // C layout: M = r + 8*seg
      sA[(r + 8 * seg) * 64 + t * 16 + mrow]      = (_Float16)lrelu(c0[r]);
      sA[(r + 8 * seg + 16) * 64 + t * 16 + mrow] = (_Float16)lrelu(c1[r]);
    }
  }
}

__global__ __launch_bounds__(256) void hashnerf_fused_kernel(
    const float* __restrict__ X,  const float* __restrict__ HT,
    const float* __restrict__ W1, const float* __restrict__ B1,
    const float* __restrict__ W2, const float* __restrict__ B2,
    const float* __restrict__ W3, const float* __restrict__ B3,
    const float* __restrict__ W4, const float* __restrict__ B4,
    float* __restrict__ out) {
  // Transposed f16 weights [n][k]; W4 padded to N=16.
  __shared__ __attribute__((aligned(16))) _Float16 sW1t[64 * 32];
  __shared__ __attribute__((aligned(16))) _Float16 sW2t[64 * 64];
  __shared__ __attribute__((aligned(16))) _Float16 sW3t[64 * 64];
  __shared__ __attribute__((aligned(16))) _Float16 sW4t[16 * 64];
  __shared__ float sB1[64], sB2[64], sB3[64], sB4[16];
  __shared__ float sHT[16];  // hash_table[h in 0..1][v in 0..3][f in 0..1]
  __shared__ __attribute__((aligned(16))) _Float16 sAct[8 * 32 * 64];

  const int tid = threadIdx.x;

  // ---- Stage weights (fp32 -> fp16, transposed) once per block ----
  for (int i = tid; i < 64 * 32; i += 256) {
    int k = i >> 6, n = i & 63;
    sW1t[n * 32 + k] = (_Float16)W1[k * 64 + n];
  }
  for (int i = tid; i < 64 * 64; i += 256) {
    int k = i >> 6, n = i & 63;
    sW2t[n * 64 + k] = (_Float16)W2[k * 64 + n];
    sW3t[n * 64 + k] = (_Float16)W3[k * 64 + n];
  }
  for (int i = tid; i < 16 * 64; i += 256) {
    int n = i >> 6, k = i & 63;
    sW4t[n * 64 + k] = (n < 3) ? (_Float16)W4[k * 3 + n] : (_Float16)0.f;
  }
  if (tid < 64) { sB1[tid] = B1[tid]; sB2[tid] = B2[tid]; sB3[tid] = B3[tid]; }
  if (tid < 16) {
    sB4[tid] = (tid < 3) ? B4[tid] : 0.f;
    // hash_table[h][v][f] at h*T*F + v*F + f; tid = h*8 + v*2 + f
    sHT[tid] = HT[(tid >> 3) * (T_SIZE * 2) + (tid & 7)];
  }
  __syncthreads();

  const int wave = tid >> 5;
  const int lane = tid & 31;
  const int mrow = lane & 15;   // WMMA row within an M-tile
  const int seg  = lane >> 4;   // lane half (K/M sub-range per ISA layout)
  _Float16* sA = sAct + wave * (32 * 64);

  const int pointBase = blockIdx.x * PTS_PER_BLOCK + wave * PTS_PER_WAVE;

  // ---- Hash-grid encoding: one lane per point, all 16 levels ----
  // N_VALUES = floor(16 * (4^(1/15))^l)
  const float NV[16] = {16.f, 17.f, 19.f, 21.f, 23.f, 25.f, 27.f, 30.f,
                        33.f, 36.f, 40.f, 44.f, 48.f, 53.f, 58.f, 64.f};
  {
    const int p = pointBase + lane;
    const float x0 = X[2 * p];
    const float x1 = X[2 * p + 1];
#pragma unroll
    for (int l = 0; l < 16; ++l) {
      const float nv = NV[l];
      const float xs = x0 * nv, ys = x1 * nv;
      const float fx = xs - floorf(xs), fy = ys - floorf(ys);
      const unsigned hx = ((unsigned)xs) & 1u;  // primes {1, odd} -> hash = LSB
      const unsigned hy = ((unsigned)ys) & 1u;
      const unsigned h3 = hx ^ hy;
      const float cx = 1.f - fx, cy = 1.f - fy;
      const float w0 = cx * cy, w1 = cx * fy, w2 = fx * cy, w3 = fx * fy;
      const float f0 = w0 * sHT[0] + w1 * sHT[hy * 8 + 2] +
                       w2 * sHT[hx * 8 + 4] + w3 * sHT[h3 * 8 + 6];
      const float f1 = w0 * sHT[1] + w1 * sHT[hy * 8 + 3] +
                       w2 * sHT[hx * 8 + 5] + w3 * sHT[h3 * 8 + 7];
      sA[lane * 64 + 2 * l]     = (_Float16)f0;
      sA[lane * 64 + 2 * l + 1] = (_Float16)f1;
    }
  }
  lds_fence();

  // ---- Layer 1: enc(32x32) @ W1(32x64) + b1, leaky relu ----
  {
    const _Float16* aRow0 = sA + mrow * 64;
    const _Float16* aRow1 = sA + (mrow + 16) * 64;
    const int kb = seg * 8;
    v16h a0 = frag_from_lds(aRow0 + kb, aRow0 + kb + 16);  // K=0..31
    v16h a1 = frag_from_lds(aRow1 + kb, aRow1 + kb + 16);
#pragma unroll
    for (int t = 0; t < 4; ++t) {
      const _Float16* bp = sW1t + (t * 16 + mrow) * 32 + seg * 8;
      v16h b = frag_from_lds(bp, bp + 16);
      const float bias = sB1[t * 16 + mrow];
      v8f c0 = splat8(bias);
      v8f c1 = splat8(bias);
      c0 = wmma_f16(a0, b, c0);
      c1 = wmma_f16(a1, b, c1);
#pragma unroll
      for (int r = 0; r < 8; ++r) {
        sA[(r + 8 * seg) * 64 + t * 16 + mrow]      = (_Float16)lrelu(c0[r]);
        sA[(r + 8 * seg + 16) * 64 + t * 16 + mrow] = (_Float16)lrelu(c1[r]);
      }
    }
  }
  lds_fence();

  // ---- Layers 2 & 3: 64 -> 64, leaky relu ----
  mlp_layer64(sA, sW2t, sB2, mrow, seg);
  lds_fence();
  mlp_layer64(sA, sW3t, sB3, mrow, seg);
  lds_fence();

  // ---- Layer 4: 64 -> 3 (padded to 16), relu, write out ----
  {
    const _Float16* aRow0 = sA + mrow * 64;
    const _Float16* aRow1 = sA + (mrow + 16) * 64;
    const int kb = seg * 8;
    v16h a00 = frag_from_lds(aRow0 + kb,      aRow0 + kb + 16);
    v16h a01 = frag_from_lds(aRow0 + 32 + kb, aRow0 + 32 + kb + 16);
    v16h a10 = frag_from_lds(aRow1 + kb,      aRow1 + kb + 16);
    v16h a11 = frag_from_lds(aRow1 + 32 + kb, aRow1 + 32 + kb + 16);
    const _Float16* bp = sW4t + mrow * 64 + seg * 8;
    v16h b0 = frag_from_lds(bp,      bp + 16);
    v16h b1 = frag_from_lds(bp + 32, bp + 48);
    const float bias = sB4[mrow];   // zero-padded for mrow >= 3
    v8f c0 = splat8(bias);
    v8f c1 = splat8(bias);
    c0 = wmma_f16(a00, b0, c0);
    c1 = wmma_f16(a10, b0, c1);
    c0 = wmma_f16(a01, b1, c0);
    c1 = wmma_f16(a11, b1, c1);
    if (mrow < 3) {  // only 3 valid output columns
#pragma unroll
      for (int r = 0; r < 8; ++r) {
        const int pr = pointBase + r + 8 * seg;
        out[pr * 3 + mrow]        = fmaxf(c0[r], 0.f);
        out[(pr + 16) * 3 + mrow] = fmaxf(c1[r], 0.f);
      }
    }
  }
}

extern "C" void kernel_launch(void* const* d_in, const int* in_sizes, int n_in,
                              void* d_out, int out_size, void* d_ws, size_t ws_size,
                              hipStream_t stream) {
  const float* X  = (const float*)d_in[0];
  const float* HT = (const float*)d_in[1];
  const float* W1 = (const float*)d_in[2];
  const float* b1 = (const float*)d_in[3];
  const float* W2 = (const float*)d_in[4];
  const float* b2 = (const float*)d_in[5];
  const float* W3 = (const float*)d_in[6];
  const float* b3 = (const float*)d_in[7];
  const float* W4 = (const float*)d_in[8];
  const float* b4 = (const float*)d_in[9];
  float* out = (float*)d_out;

  const int nPoints = in_sizes[0] / 2;               // 524288
  const int grid = nPoints / PTS_PER_BLOCK;          // 2048 blocks, 8 waves each
  hashnerf_fused_kernel<<<grid, 256, 0, stream>>>(X, HT, W1, b1, W2, b2,
                                                  W3, b3, W4, b4, out);
}
